// DecoderWithAttention_8899172238011
// MI455X (gfx1250) — matrix operand order repaired and checked by reading, not compiled, not run on hardware
//
#include <hip/hip_runtime.h>
#include <hip/hip_bf16.h>

// ---- problem constants ----
#define B_ 64
#define P_ 196
#define H_ 512
#define E_ 512
#define A_ 512
#define V_ 30000
#define L_ 21
#define T_ 20

typedef __attribute__((ext_vector_type(16))) __bf16 v16bf;
typedef __attribute__((ext_vector_type(8)))  __bf16 v8bf;
typedef __attribute__((ext_vector_type(8)))  float  v8f;

__device__ __forceinline__ float sigm(float x) { return 1.0f / (1.0f + __expf(-x)); }

// ---------------------------------------------------------------------------
// Generic WMMA GEMM:  Out[m, n] = act( X[MxK]bf16 @ W[NxK]bf16^T + bias[n]
//                                      + add[m*N+n] ) * mask(m)
// One 16x16 tile per wave, K-loop in chunks of 32 (v_wmma_f32_16x16x32_bf16).
// M, N multiples of 16; K multiple of 32.
// ---------------------------------------------------------------------------
__global__ __launch_bounds__(128)
void gemm_bt_wmma(const __bf16* __restrict__ X, const __bf16* __restrict__ W,
                  const float* __restrict__ bias, const float* __restrict__ add,
                  float* __restrict__ Out, int M, int N, int K,
                  long out_stride, int act, const int* __restrict__ lengths, int t)
{
    const int tiles_n = N >> 4;
    const int tiles_m = M >> 4;
    const int tile = blockIdx.x * 4 + (threadIdx.x >> 5);   // 4 waves / block
    if (tile >= tiles_m * tiles_n) return;                  // wave-uniform branch
    const int tm = tile / tiles_n, tn = tile % tiles_n;
    const int lane = threadIdx.x & 31;
    const int half = (lane < 16) ? 0 : 1;
    const int m0 = tm << 4, n0 = tn << 4;
    const int mrow = m0 + (lane & 15);   // A-matrix: this lane carries row mrow
    const int ncol = n0 + (lane & 15);   // B-matrix: this lane carries col ncol

    const __bf16* xrow = X + (size_t)mrow * K;
    const __bf16* wrow = W + (size_t)ncol * K;

    v8f acc = {};
    for (int k0 = 0; k0 < K; k0 += 32) {
        // A 16x32 bf16: lanes 0-15 hold K k0+[0..7],[16..23]; lanes 16-31 +8
        const int ka = k0 + half * 8;
        v8bf a_lo = *(const v8bf*)(xrow + ka);
        v8bf a_hi = *(const v8bf*)(xrow + ka + 16);
        // B 32x16 bf16: lanes 0-15 hold K k0+[0..7],[16..23]; lanes 16-31 +8
        const int kb = k0 + half * 8;
        v8bf b_lo = *(const v8bf*)(wrow + kb);
        v8bf b_hi = *(const v8bf*)(wrow + kb + 16);
        v16bf a, b;
#pragma unroll
        for (int i = 0; i < 8; ++i) {
            a[i] = a_lo[i]; a[8 + i] = a_hi[i];
            b[i] = b_lo[i]; b[8 + i] = b_hi[i];
        }
        acc = __builtin_amdgcn_wmma_f32_16x16x32_bf16(
            /*neg_a=*/false, a, /*neg_b=*/false, b,
            /*c_mod=*/(short)0, acc, /*reuse_a=*/false, /*reuse_b=*/false);
    }

    // C/D layout: VGPR r -> M = m0 + r (+8 for upper lane half); N = ncol
    const int mbase = m0 + half * 8;
#pragma unroll
    for (int r = 0; r < 8; ++r) {
        const int m = mbase + r;
        float v = acc[r];
        if (bias) v += bias[ncol];
        if (add)  v += add[(size_t)m * N + ncol];
        if (act == 1) v = sigm(v);
        if (lengths) v *= (lengths[m] - 1 > t) ? 1.0f : 0.0f;
        Out[(size_t)m * out_stride + ncol] = v;
    }
}

// ---------------------------------------------------------------------------
__global__ void f32_to_bf16(const float* __restrict__ in, __bf16* __restrict__ out, long n)
{
    long i = (long)blockIdx.x * blockDim.x + threadIdx.x;
    if (i < n) out[i] = (__bf16)in[i];
}

// h0 = c0 = mean_p encoder_out[b,p,:]
__global__ void init_h0(const float* __restrict__ enc, float* __restrict__ h,
                        float* __restrict__ c, __bf16* __restrict__ h_bf)
{
    int idx = blockIdx.x * blockDim.x + threadIdx.x;
    if (idx >= B_ * H_) return;
    int b = idx / H_, j = idx % H_;
    const float* p = enc + (size_t)b * P_ * H_ + j;
    float s = 0.0f;
    for (int pp = 0; pp < P_; ++pp) s += p[(size_t)pp * H_];
    s *= (1.0f / (float)P_);
    h[idx] = s; c[idx] = s; h_bf[idx] = (__bf16)s;
}

// Attention for one step: scores -> softmax -> gated weighted enc sum,
// builds x = [emb_t, gate*awe] in bf16, writes masked alpha to output.
__global__ __launch_bounds__(256)
void attention_step(const float* __restrict__ enc, const float* __restrict__ att1,
                    const float* __restrict__ att2, const float* __restrict__ gate,
                    const float* __restrict__ W_fa, const float* __restrict__ b_fa,
                    const float* __restrict__ emb, const int* __restrict__ captions,
                    const int* __restrict__ lengths,
                    __bf16* __restrict__ x_bf, float* __restrict__ alphas_out, int t)
{
    __shared__ float s_att2[A_];
    __shared__ float s_wfa[A_];
    __shared__ float s_e[P_];
    __shared__ float red[256];
    const int b = blockIdx.x, tid = threadIdx.x;

    for (int a = tid; a < A_; a += 256) {
        s_att2[a] = att2[(size_t)b * A_ + a];
        s_wfa[a]  = W_fa[a];
    }
    __syncthreads();

    // e[p] = relu(att1 + att2) . W_fa + b_fa
    if (tid < P_) {
        const float* row = att1 + ((size_t)b * P_ + tid) * A_;
        float s = 0.0f;
        for (int a = 0; a < A_; ++a) {
            float v = row[a] + s_att2[a];
            v = v > 0.0f ? v : 0.0f;
            s += v * s_wfa[a];
        }
        s_e[tid] = s + b_fa[0];
    }
    __syncthreads();

    // softmax over P
    float m = -3.4e38f;
    for (int p = tid; p < P_; p += 256) m = fmaxf(m, s_e[p]);
    red[tid] = m; __syncthreads();
    for (int s = 128; s > 0; s >>= 1) { if (tid < s) red[tid] = fmaxf(red[tid], red[tid + s]); __syncthreads(); }
    const float mx = red[0]; __syncthreads();
    float sum = 0.0f;
    for (int p = tid; p < P_; p += 256) { float e = __expf(s_e[p] - mx); s_e[p] = e; sum += e; }
    red[tid] = sum; __syncthreads();
    for (int s = 128; s > 0; s >>= 1) { if (tid < s) red[tid] += red[tid + s]; __syncthreads(); }
    const float inv = 1.0f / red[0]; __syncthreads();
    for (int p = tid; p < P_; p += 256) s_e[p] *= inv;   // alpha
    __syncthreads();

    // awe[h] = gate[h] * sum_p alpha[p] * enc[b,p,h]  -> x_bf[:, E: ]
    for (int hh = tid; hh < H_; hh += 256) {
        const float* col = enc + (size_t)b * P_ * H_ + hh;
        float s = 0.0f;
        for (int p = 0; p < P_; ++p) s += s_e[p] * col[(size_t)p * H_];
        s *= gate[(size_t)b * H_ + hh];
        x_bf[(size_t)b * (E_ + H_) + E_ + hh] = (__bf16)s;
    }
    // emb lookup -> x_bf[:, :E]
    const int cap = captions[b * L_ + t];
    for (int e = tid; e < E_; e += 256)
        x_bf[(size_t)b * (E_ + H_) + e] = (__bf16)emb[(size_t)cap * E_ + e];

    // masked alpha output
    const float mask = (lengths[b] - 1 > t) ? 1.0f : 0.0f;
    for (int p = tid; p < P_; p += 256)
        alphas_out[((size_t)b * T_ + t) * P_ + p] = s_e[p] * mask;
}

// LSTM pointwise: gates (B x 4H, order i,f,g,o) + c -> h,c ; also h in bf16
__global__ void lstm_point(const float* __restrict__ gates, float* __restrict__ h,
                           float* __restrict__ c, __bf16* __restrict__ h_bf)
{
    int idx = blockIdx.x * blockDim.x + threadIdx.x;
    if (idx >= B_ * H_) return;
    int b = idx / H_, j = idx % H_;
    const float* g = gates + (size_t)b * 4 * H_;
    float ig = sigm(g[j]);
    float fg = sigm(g[H_ + j]);
    float gg = tanhf(g[2 * H_ + j]);
    float og = sigm(g[3 * H_ + j]);
    float cn = fg * c[idx] + ig * gg;
    float hn = og * tanhf(cn);
    c[idx] = cn; h[idx] = hn; h_bf[idx] = (__bf16)hn;
}

// ---------------------------------------------------------------------------
static inline size_t align256(size_t x) { return (x + 255) & ~(size_t)255; }

extern "C" void kernel_launch(void* const* d_in, const int* in_sizes, int n_in,
                              void* d_out, int out_size, void* d_ws, size_t ws_size,
                              hipStream_t stream)
{
    (void)in_sizes; (void)n_in; (void)out_size; (void)ws_size;

    const float* enc      = (const float*)d_in[0];
    const int*   captions = (const int*)  d_in[1];
    const int*   lengths  = (const int*)  d_in[2];
    const float* emb      = (const float*)d_in[3];
    const float* W_ea = (const float*)d_in[4];  const float* b_ea = (const float*)d_in[5];
    const float* W_da = (const float*)d_in[6];  const float* b_da = (const float*)d_in[7];
    const float* W_fa = (const float*)d_in[8];  const float* b_fa = (const float*)d_in[9];
    const float* W_fb = (const float*)d_in[10]; const float* b_fb = (const float*)d_in[11];
    const float* W_ih = (const float*)d_in[12]; const float* W_hh = (const float*)d_in[13];
    const float* b_ih = (const float*)d_in[14]; const float* b_hh = (const float*)d_in[15];
    const float* W_fc = (const float*)d_in[16]; const float* b_fc = (const float*)d_in[17];

    float* preds_out  = (float*)d_out;                               // (B, T, V)
    float* alphas_out = (float*)d_out + (size_t)B_ * T_ * V_;        // (B, T, P)

    // ---- carve workspace ----
    char* ws = (char*)d_ws;
    size_t off = 0;
    auto carve = [&](size_t bytes) -> char* {
        char* p = ws + off; off = align256(off + bytes); return p;
    };
    __bf16* enc_bf = (__bf16*)carve((size_t)B_ * P_ * H_ * 2);
    __bf16* Wea_bf = (__bf16*)carve((size_t)A_ * H_ * 2);
    __bf16* Wda_bf = (__bf16*)carve((size_t)A_ * H_ * 2);
    __bf16* Wfb_bf = (__bf16*)carve((size_t)H_ * H_ * 2);
    __bf16* Wih_bf = (__bf16*)carve((size_t)4 * H_ * (E_ + H_) * 2);
    __bf16* Whh_bf = (__bf16*)carve((size_t)4 * H_ * H_ * 2);
    __bf16* Wfc_bf = (__bf16*)carve((size_t)V_ * H_ * 2);
    float*  att1   = (float*) carve((size_t)B_ * P_ * A_ * 4);
    float*  h      = (float*) carve((size_t)B_ * H_ * 4);
    float*  c      = (float*) carve((size_t)B_ * H_ * 4);
    __bf16* h_bf   = (__bf16*)carve((size_t)B_ * H_ * 2);
    float*  att2   = (float*) carve((size_t)B_ * A_ * 4);
    float*  gate   = (float*) carve((size_t)B_ * H_ * 4);
    float*  hWhh   = (float*) carve((size_t)B_ * 4 * H_ * 4);
    __bf16* x_bf   = (__bf16*)carve((size_t)B_ * (E_ + H_) * 2);
    float*  gates  = (float*) carve((size_t)B_ * 4 * H_ * 4);

    auto cvt = [&](const float* src, __bf16* dst, long n) {
        f32_to_bf16<<<(unsigned)((n + 255) / 256), 256, 0, stream>>>(src, dst, n);
    };
    // bf16 copies of everything that feeds WMMA
    cvt(enc,  enc_bf, (long)B_ * P_ * H_);
    cvt(W_ea, Wea_bf, (long)A_ * H_);
    cvt(W_da, Wda_bf, (long)A_ * H_);
    cvt(W_fb, Wfb_bf, (long)H_ * H_);
    cvt(W_ih, Wih_bf, (long)4 * H_ * (E_ + H_));
    cvt(W_hh, Whh_bf, (long)4 * H_ * H_);
    cvt(W_fc, Wfc_bf, (long)V_ * H_);

    // h0 = c0 = mean over P
    init_h0<<<(B_ * H_ + 255) / 256, 256, 0, stream>>>(enc, h, c, h_bf);

    auto gemm_grid = [](int M, int N) -> unsigned {
        return (unsigned)(((M >> 4) * (N >> 4) + 3) / 4);
    };

    // att1 = enc @ W_ea^T + b_ea   (B*P x A)
    gemm_bt_wmma<<<gemm_grid(B_ * P_, A_), 128, 0, stream>>>(
        enc_bf, Wea_bf, b_ea, nullptr, att1, B_ * P_, A_, H_, (long)A_, 0, nullptr, 0);

    for (int t = 0; t < T_; ++t) {
        // att2 = h @ W_da^T + b_da
        gemm_bt_wmma<<<gemm_grid(B_, A_), 128, 0, stream>>>(
            h_bf, Wda_bf, b_da, nullptr, att2, B_, A_, H_, (long)A_, 0, nullptr, 0);
        // gate = sigmoid(h @ W_fb^T + b_fb)
        gemm_bt_wmma<<<gemm_grid(B_, H_), 128, 0, stream>>>(
            h_bf, Wfb_bf, b_fb, nullptr, gate, B_, H_, H_, (long)H_, 1, nullptr, 0);
        // hWhh = h @ W_hh^T + b_hh
        gemm_bt_wmma<<<gemm_grid(B_, 4 * H_), 128, 0, stream>>>(
            h_bf, Whh_bf, b_hh, nullptr, hWhh, B_, 4 * H_, H_, (long)(4 * H_), 0, nullptr, 0);
        // attention + build x = [emb_t, gate*awe]
        attention_step<<<B_, 256, 0, stream>>>(
            enc, att1, att2, gate, W_fa, b_fa, emb, captions, lengths,
            x_bf, alphas_out, t);
        // gates = x @ W_ih^T + b_ih + hWhh
        gemm_bt_wmma<<<gemm_grid(B_, 4 * H_), 128, 0, stream>>>(
            x_bf, Wih_bf, b_ih, hWhh, gates, B_, 4 * H_, E_ + H_, (long)(4 * H_), 0, nullptr, 0);
        // LSTM pointwise -> h, c, h_bf
        lstm_point<<<(B_ * H_ + 255) / 256, 256, 0, stream>>>(gates, h, c, h_bf);
        // preds[:, t, :] = mask * (h_new @ W_fc^T + b_fc), written in-place to d_out
        gemm_bt_wmma<<<gemm_grid(B_, V_), 128, 0, stream>>>(
            h_bf, Wfc_bf, b_fc, nullptr, preds_out + (size_t)t * V_,
            B_, V_, H_, (long)T_ * V_, 0, lengths, t);
    }
}
